// Quant4Linear_normal_eora_dummy_32762010534273
// MI455X (gfx1250) — matrix operand order repaired and checked by reading, not compiled, not run on hardware
//
#include <hip/hip_runtime.h>

// Problem constants (match reference)
#define KDIM 8192
#define NDIM 28672
#define RANK 64
#define K8   (KDIM / 8)      // 1024 packed int32 rows
#define NBLK (NDIM / 256)    // 112 blocks of 256 columns
#define RBLK 64              // reduce blocks, 128 k-rows each

typedef __attribute__((ext_vector_type(16))) _Float16 v16h;
typedef __attribute__((ext_vector_type(8)))  float    v8f;

// ---------------------------------------------------------------------------
// Kernel 1: deterministic partial reduction for down_proj = x @ down (K x 64)
// ws layout: ws[0] = sum(x); ws[1 + b*64 + r] = block-b partial of down_proj[r]
// ---------------------------------------------------------------------------
__global__ void k_reduce(const float* __restrict__ x,
                         const _Float16* __restrict__ down,
                         float* __restrict__ ws) {
  __shared__ float red[256];
  const int b = blockIdx.x;
  const int t = threadIdx.x;
  const int r    = t & 63;   // rank index
  const int part = t >> 6;   // 4 k-subchunks of 32 rows
  const int k0 = b * 128 + part * 32;

  float acc = 0.f;
  #pragma unroll 8
  for (int i = 0; i < 32; ++i) {
    const int k = k0 + i;                       // k uniform per wave -> x via SMEM
    acc = fmaf(x[k], (float)down[k * RANK + r], acc);
  }
  red[t] = acc;
  __syncthreads();
  if (t < 64) {
    ws[1 + b * 64 + t] = red[t] + red[t + 64] + red[t + 128] + red[t + 192];
  }

  // Block 0 additionally computes sum(x) (deterministic tree reduce)
  if (b == 0) {
    __syncthreads();
    float xa = 0.f;
    for (int k = t; k < KDIM; k += 256) xa += x[k];
    red[t] = xa;
    for (int off = 128; off > 0; off >>= 1) {
      __syncthreads();
      if (t < off) red[t] += red[t + off];
    }
    __syncthreads();
    if (t == 0) ws[0] = red[0];
  }
}

// ---------------------------------------------------------------------------
// Kernel 2: out[n] = bias[n] + corr[n] - zeros[n]*sum(x)
// corr computed with v_wmma_f32_16x16x32_f16: A = down_proj broadcast (row 0),
// B = up[ k-tile, 16-column tile ] in contiguous k order (K = 16h + e). Two k-steps cover RANK=64.
// ---------------------------------------------------------------------------
__global__ void k_base(const float* __restrict__ ws,
                       const _Float16* __restrict__ up,
                       const float* __restrict__ bias,
                       const float* __restrict__ zeros,
                       float* __restrict__ out) {
  __shared__ _Float16 dp[RANK];
  __shared__ float corr[256];
  const int t = threadIdx.x;

  // Fold the 64 deterministic partials into down_proj (fp16, like reference)
  if (t < 64) {
    float s = 0.f;
    #pragma unroll 8
    for (int b = 0; b < RBLK; ++b) s += ws[1 + b * 64 + t];
    dp[t] = (_Float16)s;
  }
  __syncthreads();

  const int lane = t & 31;
  const int wave = t >> 5;

  // A-matrix (16x32 f16): row 0 = down_proj chunk, rows 1..15 = 0.
  // Lanes 0-15 hold K=0..15, lanes 16-31 hold K=16..31 (matches B);
  // element e -> K = e (+16 for upper lanes).
  const bool act  = (lane == 0) || (lane == 16);
  const int  koff = (lane >= 16) ? 16 : 0;
  v16h a0, a1;
  #pragma unroll
  for (int e = 0; e < 16; ++e) {
    const int k = e + koff;
    a0[e] = act ? dp[k]      : (_Float16)0.f;
    a1[e] = act ? dp[32 + k] : (_Float16)0.f;
  }

  // Each of the 8 waves computes two 16-column tiles (block covers 256 cols)
  #pragma unroll
  for (int i = 0; i < 2; ++i) {
    const int tile = wave * 2 + i;
    const int col  = blockIdx.x * 256 + tile * 16 + (lane & 15);
    const int krow = (lane >> 4) * 16;   // B layout: lanes 0-15 K=0..15, 16-31 K=16..31
    v16h b0, b1;
    #pragma unroll
    for (int e = 0; e < 16; ++e) {
      b0[e] = up[(krow + e) * NDIM + col];
      b1[e] = up[(32 + krow + e) * NDIM + col];
    }
    v8f c = {};
    c = __builtin_amdgcn_wmma_f32_16x16x32_f16(false, a0, false, b0,
                                               (short)0, c, false, false);
    c = __builtin_amdgcn_wmma_f32_16x16x32_f16(false, a1, false, b1,
                                               (short)0, c, false, false);
    // D layout: VGPR0, lanes 0-15 = (M=0, N=lane) -> corr for this tile
    if (lane < 16) corr[tile * 16 + lane] = c[0];
  }
  __syncthreads();

  const float xsum = ws[0];
  const int n = blockIdx.x * 256 + t;
  out[n] = bias[n] + corr[t] - zeros[n] * xsum;
}

// ---------------------------------------------------------------------------
// Kernel 3: bandwidth-bound int4 dequant GEMV.
// 1 thread = 1 output column; wave streams 128B contiguous of qweight per k8.
// Dequant: dual 0x0F0F0F0F masks -> v_cvt_f32_ubyte + v_fma_f32 per nibble.
// x fetched with uniform (scalar/SMEM) loads. scale/zero applied once at end.
// ---------------------------------------------------------------------------
__global__ void k_gemv(const float* __restrict__ x,
                       const int* __restrict__ qw,
                       const float* __restrict__ scales,
                       float* __restrict__ out) {
  const int n = blockIdx.x * 256 + threadIdx.x;
  float acc = 0.f;
  #pragma unroll 4
  for (int k8 = 0; k8 < K8; ++k8) {
    const unsigned q  = (unsigned)qw[k8 * NDIM + n];
    const unsigned lo = q & 0x0F0F0F0Fu;          // nibbles 0,2,4,6 in bytes
    const unsigned hi = (q >> 4) & 0x0F0F0F0Fu;   // nibbles 1,3,5,7 in bytes
    const float* xk = x + k8 * 8;                 // uniform -> scalar loads
    acc = fmaf((float)( lo        & 0xFFu), xk[0], acc);
    acc = fmaf((float)( hi        & 0xFFu), xk[1], acc);
    acc = fmaf((float)((lo >>  8) & 0xFFu), xk[2], acc);
    acc = fmaf((float)((hi >>  8) & 0xFFu), xk[3], acc);
    acc = fmaf((float)((lo >> 16) & 0xFFu), xk[4], acc);
    acc = fmaf((float)((hi >> 16) & 0xFFu), xk[5], acc);
    acc = fmaf((float)( lo >> 24),          xk[6], acc);
    acc = fmaf((float)( hi >> 24),          xk[7], acc);
  }
  out[n] += scales[n] * acc;   // base (bias+corr-zero*xsum) already in out
}

// ---------------------------------------------------------------------------
extern "C" void kernel_launch(void* const* d_in, const int* in_sizes, int n_in,
                              void* d_out, int out_size, void* d_ws, size_t ws_size,
                              hipStream_t stream) {
  (void)in_sizes; (void)n_in; (void)out_size; (void)ws_size;
  const float*    x      = (const float*)d_in[0];
  const int*      qw     = (const int*)d_in[1];
  const float*    scales = (const float*)d_in[2];
  const float*    zeros  = (const float*)d_in[3];
  const float*    bias   = (const float*)d_in[4];
  const _Float16* down   = (const _Float16*)d_in[5];
  const _Float16* up     = (const _Float16*)d_in[6];
  float* out = (float*)d_out;
  float* ws  = (float*)d_ws;   // needs (1 + 64*64) floats = 16.4 KB

  k_reduce<<<RBLK, 256, 0, stream>>>(x, down, ws);
  k_base <<<NBLK, 256, 0, stream>>>(ws, up, bias, zeros, out);
  k_gemv <<<NBLK, 256, 0, stream>>>(x, qw, scales, out);
}
